// GIN_20907900796962
// MI455X (gfx1250) — matrix operand order repaired and checked by reading, not compiled, not run on hardware
//
#include <hip/hip_runtime.h>
#include <hip/hip_bf16.h>

// GIN forward for MI455X (gfx1250, wave32, WMMA).
//   N=100000 nodes, E=1600000 edges, H=256.
// Strategy:
//   - Build CSR (by dst) once per launch -> gather-based segment_sum (no f32 atomics).
//   - agg2 first half == agg  (xc = [x,h]) -> only h needs a 2nd aggregation pass.
//   - GEMMs via v_wmma_f32_16x16x32_f16 with split-f16 (Markidis) for ~f32 accuracy:
//       y = Ahi*Bhi + Ahi*Blo + Alo*Bhi   (f32 accumulate)
//   - Weights pre-packed into B-fragment layout; per K-tile fragments staged to LDS
//     once per 64-row block via async global->LDS copies (double-buffered, ASYNCcnt).
//   - Inner loop processes 2 n-tiles at once so dependent WMMA accumulation chains
//     are interleaved with independent WMMAs (hides XDL result latency).
//   - Bias skipped: cancels exactly under batch-stat BatchNorm (and b==0 anyway).
//   - BN column sums/sumsq fused into GEMM epilogue via atomics; tiny finalize kernel;
//     in-place scale/shift + ReLU.

#define USE_ASYNC_STAGE 1

typedef _Float16 f16;
typedef __attribute__((ext_vector_type(16))) _Float16 v16h;
typedef __attribute__((ext_vector_type(8)))  float    v8f;

#define HDIM 256

// ------------------------------ CSR build ------------------------------

__global__ void k_degree(const int* __restrict__ dst, int* __restrict__ deg, int E) {
  for (int e = blockIdx.x * blockDim.x + threadIdx.x; e < E; e += gridDim.x * blockDim.x)
    atomicAdd(&deg[dst[e]], 1);
}

// Single-workgroup chunked exclusive scan (N ~ 100K -> ~98 chunks of 1024).
__global__ void k_scan(const int* __restrict__ deg, int* __restrict__ ptr, int n) {
  __shared__ int sm[1024];
  __shared__ int carry;
  int tid = threadIdx.x;
  if (tid == 0) carry = 0;
  __syncthreads();
  for (int base = 0; base < n; base += 1024) {
    int v = (base + tid < n) ? deg[base + tid] : 0;
    sm[tid] = v;
    __syncthreads();
    for (int off = 1; off < 1024; off <<= 1) {
      int t = (tid >= off) ? sm[tid - off] : 0;
      __syncthreads();
      sm[tid] += t;
      __syncthreads();
    }
    if (base + tid < n) ptr[base + tid] = carry + sm[tid] - v;   // exclusive
    __syncthreads();
    if (tid == 0) carry += sm[1023];
    __syncthreads();
  }
  if (tid == 0) ptr[n] = carry;
}

__global__ void k_fill(const int* __restrict__ src, const int* __restrict__ dst,
                       int* __restrict__ cur, int* __restrict__ idx, int E) {
  for (int e = blockIdx.x * blockDim.x + threadIdx.x; e < E; e += gridDim.x * blockDim.x) {
    int p = atomicAdd(&cur[dst[e]], 1);
    idx[p] = src[e];
  }
}

// ------------------------ gather aggregation ---------------------------
// out[i] = X[i] + sum_{j in in(i)} X[j]    (one wave per node, 8 f32/lane)
__global__ void k_gather_sum(const float* __restrict__ X, const int* __restrict__ ptr,
                             const int* __restrict__ idx, float* __restrict__ out, int n) {
  int node = (blockIdx.x * blockDim.x + threadIdx.x) >> 5;
  int lane = threadIdx.x & 31;
  if (node >= n) return;
  const float* xs = X + (size_t)node * HDIM;
  float4 a0 = *(const float4*)(xs + 4 * lane);
  float4 a1 = *(const float4*)(xs + 128 + 4 * lane);
  int b = ptr[node], e = ptr[node + 1];
  for (; b < e; ++b) {
    const float* xj = X + (size_t)idx[b] * HDIM;
    float4 b0 = *(const float4*)(xj + 4 * lane);
    float4 b1 = *(const float4*)(xj + 128 + 4 * lane);
    a0.x += b0.x; a0.y += b0.y; a0.z += b0.z; a0.w += b0.w;
    a1.x += b1.x; a1.y += b1.y; a1.z += b1.z; a1.w += b1.w;
  }
  float* o = out + (size_t)node * HDIM;
  *(float4*)(o + 4 * lane) = a0;
  *(float4*)(o + 128 + 4 * lane) = a1;
}

// -------------------- weight packing into B-fragments -------------------
// B (32x16 f16) fragment in this code's contiguous k order (same as A): lane = (n&15) + 16*((k&31)>>4), element i <-> k&15.
// Packed per K-tile as one contiguous 32KB block: [kt]{ hi[16nt][32lane][16], lo[...] }.
__global__ void k_pack_w(const float* __restrict__ W, f16* __restrict__ pw, int K) {
  int total = K * HDIM;
  for (int e = blockIdx.x * blockDim.x + threadIdx.x; e < total; e += gridDim.x * blockDim.x) {
    int k = e >> 8;
    int n = e & 255;
    float w = W[e];
    f16 h = (f16)w;
    f16 l = (f16)(w - (float)h);
    int kt = k >> 5, r = k & 31, g = r >> 4, i = r & 15;
    int nt = n >> 4, ln = (n & 15) + 16 * g;
    size_t base = (size_t)kt * 16384;               // 2 * 8192 f16 per K-tile
    size_t fo   = (size_t)(nt * 32 + ln) * 16 + i;
    pw[base + fo]        = h;                       // hi section
    pw[base + 8192 + fo] = l;                       // lo section
  }
}

// ------------------------------- GEMM ----------------------------------
// Y[N,256] = A[N,K] * W[K,256];   K = KT*32 (KT=8 -> A0 only, KT=16 -> A0|A1 halves).
// Block = 256 thr = 8 waves; block tile 64(M) x 256(N); wave = 16(M) x 128(N) (8 nt).
// B fragments double-buffered in LDS (async global->LDS), one barrier per K-tile.
// Fused epilogue: per-column sum / sumsq atomics for BatchNorm batch stats.

__device__ __forceinline__ void stage_tile(const f16* __restrict__ src, f16* dstLds, int tid) {
#if USE_ASYNC_STAGE
  const char* s = (const char*)src;
#pragma unroll
  for (int c = 0; c < 8; ++c) {                      // 8 * 256thr * 16B = 32KB
    unsigned off = (unsigned)((c * 256 + tid) * 16);
    unsigned lds = (unsigned)(uintptr_t)((char*)dstLds + off);
    asm volatile("global_load_async_to_lds_b128 %0, %1, %2 offset:0"
                 :: "v"(lds), "v"(off), "s"(s) : "memory");
  }
#else
  const float4* s = (const float4*)src;
  float4* d = (float4*)dstLds;
#pragma unroll
  for (int c = 0; c < 8; ++c) d[c * 256 + tid] = s[c * 256 + tid];
#endif
}

__global__ __launch_bounds__(256) void k_gemm(
    const float* __restrict__ A0, const float* __restrict__ A1,
    const f16* __restrict__ P, float* __restrict__ Y,
    float* __restrict__ sum, float* __restrict__ sumsq, int KT, int Nrows) {
  __shared__ __align__(16) f16 sB[2][16384];         // 2 x 32KB double buffer
  const int tid  = threadIdx.x;
  const int lane = tid & 31;
  const int wave = tid >> 5;
  const int mt   = wave & 3;                         // 4 row-tiles of 16
  const int ntg  = wave >> 2;                        // 2 groups of 8 n-tiles
  const int tileBase = blockIdx.x * 64 + mt * 16;
  const int m16 = lane & 15, g = lane >> 4;
  const int arow = tileBase + m16;                   // tail rows read in-ws garbage (safe)

  v8f acc[8] = {};

  stage_tile(P, &sB[0][0], tid);                     // prologue: kt = 0 -> buf 0

  for (int kt = 0; kt < KT; ++kt) {
#if USE_ASYNC_STAGE
    asm volatile("s_wait_asynccnt 0x0" ::: "memory");  // own async chunks done
#endif
    __syncthreads();                                 // all waves' chunks done; prev reads done
    if (kt + 1 < KT)
      stage_tile(P + (size_t)(kt + 1) * 16384, &sB[(kt + 1) & 1][0], tid);

    const float* A = (kt < 8) ? A0 : A1;
    int kc = (kt & 7) * 32;
    // A fragment (16x32 f16): lane holds row m16, k = kc+16g+[0..16)
    const float* p = A + (size_t)arow * HDIM + kc + 16 * g;
    float4 f0 = *(const float4*)(p);
    float4 f1 = *(const float4*)(p + 4);
    float4 f2 = *(const float4*)(p + 8);
    float4 f3 = *(const float4*)(p + 12);
    float t16[16] = {f0.x, f0.y, f0.z, f0.w, f1.x, f1.y, f1.z, f1.w,
                     f2.x, f2.y, f2.z, f2.w, f3.x, f3.y, f3.z, f3.w};
    v16h ahi, alo;
#pragma unroll
    for (int i = 0; i < 16; ++i) {
      f16 h = (f16)t16[i];
      ahi[i] = h;
      alo[i] = (f16)(t16[i] - (float)h);
    }
    const f16* Bb = &sB[kt & 1][0];
    // 2 n-tiles per iteration: interleave the two accumulation chains so each
    // dependent WMMA is separated by an independent one (hide XDL latency).
#pragma unroll
    for (int t = 0; t < 8; t += 2) {
      int nt0 = ntg * 8 + t;
      int nt1 = nt0 + 1;
      v16h bhi0 = *(const v16h*)(Bb + (size_t)(nt0 * 32 + lane) * 16);
      v16h blo0 = *(const v16h*)(Bb + 8192 + (size_t)(nt0 * 32 + lane) * 16);
      v16h bhi1 = *(const v16h*)(Bb + (size_t)(nt1 * 32 + lane) * 16);
      v16h blo1 = *(const v16h*)(Bb + 8192 + (size_t)(nt1 * 32 + lane) * 16);
      acc[t]     = __builtin_amdgcn_wmma_f32_16x16x32_f16(false, ahi, false, bhi0,
                                                          (short)0, acc[t], false, false);
      acc[t + 1] = __builtin_amdgcn_wmma_f32_16x16x32_f16(false, ahi, false, bhi1,
                                                          (short)0, acc[t + 1], false, false);
      acc[t]     = __builtin_amdgcn_wmma_f32_16x16x32_f16(false, ahi, false, blo0,
                                                          (short)0, acc[t], false, false);
      acc[t + 1] = __builtin_amdgcn_wmma_f32_16x16x32_f16(false, ahi, false, blo1,
                                                          (short)0, acc[t + 1], false, false);
      acc[t]     = __builtin_amdgcn_wmma_f32_16x16x32_f16(false, alo, false, bhi0,
                                                          (short)0, acc[t], false, false);
      acc[t + 1] = __builtin_amdgcn_wmma_f32_16x16x32_f16(false, alo, false, bhi1,
                                                          (short)0, acc[t + 1], false, false);
    }
  }

  // C/D layout: element v <-> row tileBase + v + 8g, col = nt*16 + (lane&15)
#pragma unroll
  for (int t = 0; t < 8; ++t) {
    int col = (ntg * 8 + t) * 16 + m16;
    float s = 0.f, s2 = 0.f;
#pragma unroll
    for (int v = 0; v < 8; ++v) {
      int rr = tileBase + v + 8 * g;
      if (rr < Nrows) {
        float c = acc[t][v];
        Y[(size_t)rr * HDIM + col] = c;
        s += c;
        s2 += c * c;
      }
    }
    atomicAdd(&sum[col], s);
    atomicAdd(&sumsq[col], s2);
  }
}

// --------------------------- BN finalize/apply --------------------------

__global__ void k_bn_finalize(const float* __restrict__ sum, const float* __restrict__ sumsq,
                              const float* __restrict__ gamma, const float* __restrict__ beta,
                              float* __restrict__ scale, float* __restrict__ shift, float invN) {
  int c = threadIdx.x;
  float mu  = sum[c] * invN;
  float var = sumsq[c] * invN - mu * mu;
  float sc  = gamma[c] * rsqrtf(var + 1e-5f);
  scale[c] = sc;
  shift[c] = beta[c] - mu * sc;
}

__global__ void k_bnrelu(float* __restrict__ Y, const float* __restrict__ scale,
                         const float* __restrict__ shift, long n4) {
  for (long i = blockIdx.x * (long)blockDim.x + threadIdx.x; i < n4;
       i += (long)gridDim.x * blockDim.x) {
    float4 v = ((float4*)Y)[i];
    int cb = (int)((i * 4) & 255);
    v.x = fmaxf(v.x * scale[cb + 0] + shift[cb + 0], 0.f);
    v.y = fmaxf(v.y * scale[cb + 1] + shift[cb + 1], 0.f);
    v.z = fmaxf(v.z * scale[cb + 2] + shift[cb + 2], 0.f);
    v.w = fmaxf(v.w * scale[cb + 3] + shift[cb + 3], 0.f);
    ((float4*)Y)[i] = v;
  }
}

// ------------------------------ launch ---------------------------------

extern "C" void kernel_launch(void* const* d_in, const int* in_sizes, int n_in,
                              void* d_out, int out_size, void* d_ws, size_t ws_size,
                              hipStream_t stream) {
  const int N = in_sizes[0] / HDIM;   // 100000
  const int E = in_sizes[1] / 2;      // 1600000

  const float* x    = (const float*)d_in[0];
  const int*   esrc = (const int*)d_in[1];
  const int*   edst = esrc + E;
  const float* W1 = (const float*)d_in[2];
  const float* g1 = (const float*)d_in[4];
  const float* be1 = (const float*)d_in[5];
  const float* W2 = (const float*)d_in[6];
  const float* g2 = (const float*)d_in[8];
  const float* be2 = (const float*)d_in[9];
  const float* W3 = (const float*)d_in[10];
  const float* g3 = (const float*)d_in[12];
  const float* be3 = (const float*)d_in[13];
  const float* W4 = (const float*)d_in[14];
  const float* g4 = (const float*)d_in[16];
  const float* be4 = (const float*)d_in[17];
  const float* W5 = (const float*)d_in[18];
  const float* g5 = (const float*)d_in[20];
  const float* be5 = (const float*)d_in[21];

  // ---- workspace layout ----
  char* w = (char*)d_ws;
  size_t off = 0;
  auto take = [&](size_t bytes) -> void* {
    off = (off + 255) & ~(size_t)255;
    void* p = w + off;
    off += bytes;
    return p;
  };
  const size_t nodeBytes = (size_t)N * HDIM * sizeof(float);
  int*   deg  = (int*)take((size_t)N * 4);         // also reused as fill cursor
  int*   ptr  = (int*)take((size_t)(N + 1) * 4);
  int*   idx  = (int*)take((size_t)E * 4);
  float* tX   = (float*)take(nodeBytes);           // x + sum(x)
  float* Y    = (float*)take(nodeBytes);           // gemm out / h1 / h3
  float* Hb   = (float*)take(nodeBytes);           // h (layer2 out) / h4
  float* tH   = (float*)take(nodeBytes);           // h + sum(h)
  float* stats = (float*)take(1024 * sizeof(float));
  float* sum   = stats;
  float* sumsq = stats + 256;
  float* scale = stats + 512;
  float* shift = stats + 768;
  f16* pw1 = (f16*)take((size_t)8  * 16384 * 2);   // K=256: 8 K-tiles x 32KB
  f16* pw2 = (f16*)take((size_t)8  * 16384 * 2);
  f16* pw3 = (f16*)take((size_t)16 * 16384 * 2);   // K=512: 16 K-tiles x 32KB
  f16* pw4 = (f16*)take((size_t)8  * 16384 * 2);
  f16* pw5 = (f16*)take((size_t)8  * 16384 * 2);
  (void)ws_size; (void)n_in; (void)out_size;

  const int eBlocks   = (E + 255) / 256;
  const int gemmGrid  = (N + 63) / 64;             // 64 rows per block
  const int aggBlocks = (N * 32 + 255) / 256;      // one wave per node
  const long n4       = (long)N * 64;
  const int bnBlocks  = (int)((n4 + 255) / 256);
  const float invN    = 1.0f / (float)N;

  // ---- CSR build ----
  hipMemsetAsync(deg, 0, (size_t)N * 4, stream);
  k_degree<<<eBlocks, 256, 0, stream>>>(edst, deg, E);
  k_scan<<<1, 1024, 0, stream>>>(deg, ptr, N);
  hipMemcpyAsync(deg, ptr, (size_t)N * 4, hipMemcpyDeviceToDevice, stream);  // cursor
  k_fill<<<eBlocks, 256, 0, stream>>>(esrc, edst, deg, idx, E);

  // ---- pack weights (hi/lo f16 B-fragments, contiguous 32KB per K-tile) ----
  k_pack_w<<<256, 256, 0, stream>>>(W1, pw1, 256);
  k_pack_w<<<256, 256, 0, stream>>>(W2, pw2, 256);
  k_pack_w<<<512, 256, 0, stream>>>(W3, pw3, 512);
  k_pack_w<<<256, 256, 0, stream>>>(W4, pw4, 256);
  k_pack_w<<<256, 256, 0, stream>>>(W5, pw5, 256);

  // ---- aggregation 1: tX = x + segment_sum(x) ----
  k_gather_sum<<<aggBlocks, 256, 0, stream>>>(x, ptr, idx, tX, N);

  // ---- layer 1: Y = bnrelu(tX @ W1) ----
  hipMemsetAsync(stats, 0, 512 * sizeof(float), stream);
  k_gemm<<<gemmGrid, 256, 0, stream>>>(tX, tX, pw1, Y, sum, sumsq, 8, N);
  k_bn_finalize<<<1, 256, 0, stream>>>(sum, sumsq, g1, be1, scale, shift, invN);
  k_bnrelu<<<bnBlocks, 256, 0, stream>>>(Y, scale, shift, n4);

  // ---- layer 2: Hb = bnrelu(Y @ W2)  (== h) ----
  hipMemsetAsync(stats, 0, 512 * sizeof(float), stream);
  k_gemm<<<gemmGrid, 256, 0, stream>>>(Y, Y, pw2, Hb, sum, sumsq, 8, N);
  k_bn_finalize<<<1, 256, 0, stream>>>(sum, sumsq, g2, be2, scale, shift, invN);
  k_bnrelu<<<bnBlocks, 256, 0, stream>>>(Hb, scale, shift, n4);

  // ---- aggregation 2: tH = h + segment_sum(h)  (first half of agg2 is tX) ----
  k_gather_sum<<<aggBlocks, 256, 0, stream>>>(Hb, ptr, idx, tH, N);

  // ---- layer 3: Y = bnrelu([tX | tH] @ W3)  (K = 512) ----
  hipMemsetAsync(stats, 0, 512 * sizeof(float), stream);
  k_gemm<<<gemmGrid, 256, 0, stream>>>(tX, tH, pw3, Y, sum, sumsq, 16, N);
  k_bn_finalize<<<1, 256, 0, stream>>>(sum, sumsq, g3, be3, scale, shift, invN);
  k_bnrelu<<<bnBlocks, 256, 0, stream>>>(Y, scale, shift, n4);

  // ---- layer 4: Hb = bnrelu(Y @ W4) ----
  hipMemsetAsync(stats, 0, 512 * sizeof(float), stream);
  k_gemm<<<gemmGrid, 256, 0, stream>>>(Y, Y, pw4, Hb, sum, sumsq, 8, N);
  k_bn_finalize<<<1, 256, 0, stream>>>(sum, sumsq, g4, be4, scale, shift, invN);
  k_bnrelu<<<bnBlocks, 256, 0, stream>>>(Hb, scale, shift, n4);

  // ---- layer 5: out = bnrelu(Hb @ W5) ----
  float* out = (float*)d_out;
  hipMemsetAsync(stats, 0, 512 * sizeof(float), stream);
  k_gemm<<<gemmGrid, 256, 0, stream>>>(Hb, Hb, pw5, out, sum, sumsq, 8, N);
  k_bn_finalize<<<1, 256, 0, stream>>>(sum, sumsq, g5, be5, scale, shift, invN);
  k_bnrelu<<<bnBlocks, 256, 0, stream>>>(out, scale, shift, n4);
}
